// PerceiverAttention_21887153341108
// MI455X (gfx1250) — matrix operand order repaired and checked by reading, not compiled, not run on hardware
//
#include <hip/hip_runtime.h>

typedef unsigned short ushort_t;
typedef unsigned int uint32;
typedef unsigned long long uint64;

typedef __attribute__((ext_vector_type(16))) __bf16 v16bf;
typedef __attribute__((ext_vector_type(2)))  __bf16 v2bf;
typedef __attribute__((ext_vector_type(8)))  float  v8f;

// ---------- bf16 helpers ----------
// native fptrunc: lets LLVM pick the hw bf16 convert on gfx1250 (RNE semantics)
__device__ __forceinline__ ushort_t f2bf(float f) {
    union { __bf16 b; ushort_t u; } x;
    x.b = (__bf16)f;
    return x.u;
}
__device__ __forceinline__ float bf2f(ushort_t h) {
    return __uint_as_float(((uint32)h) << 16);
}
// pack two f32 -> one dword of 2 x bf16
__device__ __forceinline__ uint32 pack2bf(float lo, float hi) {
#if __has_builtin(__builtin_amdgcn_cvt_pk_bf16_f32)
    union { v2bf v; uint32 u; } r;
    r.v = __builtin_amdgcn_cvt_pk_bf16_f32(lo, hi);
    return r.u;
#else
    union { v2bf v; uint32 u; } r;
    r.v = v2bf{(__bf16)lo, (__bf16)hi};
    return r.u;
#endif
}

// ---------- CDNA5 async global->LDS copy (16B per lane) ----------
__device__ __forceinline__ void async_ld_b128(void* lds_dst, const void* gsrc) {
    uint32 lds = (uint32)(size_t)lds_dst;      // low 32 bits of generic ptr = LDS offset
    uint64 ga  = (uint64)(size_t)gsrc;
    asm volatile("global_load_async_to_lds_b128 %0, %1, off"
                 :: "v"(lds), "v"(ga) : "memory");
}
__device__ __forceinline__ void async_wait0() {
    asm volatile("s_wait_asynccnt 0x0" ::: "memory");
}

// ---------- WMMA fragment loaders (layouts per CDNA5 ISA) ----------
struct FragU { union { uint4 q[2]; v16bf v; }; };

// A (16x32 bf16): lane L -> row = base+(L&15); chunks [kg..kg+7], [kg+16..kg+23], kg=(L>>4)*8
__device__ __forceinline__ v16bf load_frag_a(const ushort_t* base, int pitch, int row, int lane, int koff) {
    FragU f;
    int kg = ((lane >> 4) & 1) * 8 + koff;
    const ushort_t* p = base + (row + (lane & 15)) * pitch + kg;
    f.q[0] = *(const uint4*)(p);
    f.q[1] = *(const uint4*)(p + 16);
    return f.v;
}
// B (32x16 bf16), staged K-contiguous per column: chunks [kg..kg+7], [kg+16..kg+23], kg=(L>>4)*8
__device__ __forceinline__ v16bf load_frag_b(const ushort_t* base, int pitch, int col, int lane, int koff) {
    FragU f;
    int kb = ((lane >> 4) & 1) * 8 + koff;
    const ushort_t* p = base + (col + (lane & 15)) * pitch + kb;
    f.q[0] = *(const uint4*)(p);
    f.q[1] = *(const uint4*)(p + 16);
    return f.v;
}

// ---------- fp32 -> bf16 convert (paired) ----------
__global__ __launch_bounds__(256) void cvt_bf16_kernel(const float* __restrict__ in,
                                                       ushort_t* __restrict__ out, int n) {
    int i = (blockIdx.x * 256 + threadIdx.x) * 2;
    if (i < n) {
        float2 f = *(const float2*)&in[i];
        *(uint32*)&out[i] = pack2bf(f.x, f.y);
    }
}

// ---------- LayerNorm(x) & LayerNorm(latents) -> bf16 kv_input rows (+ contiguous ln) ----------
__global__ __launch_bounds__(256) void ln_kernel(
    const float* __restrict__ x, const float* __restrict__ lat,
    const float* __restrict__ gm, const float* __restrict__ bm,
    const float* __restrict__ gl, const float* __restrict__ bl,
    ushort_t* __restrict__ kvin, ushort_t* __restrict__ lnout) {
    int r = blockIdx.x;
    int bt = r / 640, s = r % 640;
    bool isl = (s >= 576);
    const float* src = isl ? (lat + ((size_t)bt * 64 + (s - 576)) * 1024)
                           : (x   + ((size_t)bt * 576 + s) * 1024);
    const float* g = isl ? gl : gm;
    const float* b = isl ? bl : bm;
    int t = threadIdx.x, lane = t & 31, w = t >> 5;
    int c0 = t * 4;

    float4 v = *(const float4*)&src[c0];
    float sum = v.x + v.y + v.z + v.w;
    float sq  = v.x * v.x + v.y * v.y + v.z * v.z + v.w * v.w;

    __shared__ float red[16];
    for (int o = 16; o > 0; o >>= 1) { sum += __shfl_xor(sum, o, 32); sq += __shfl_xor(sq, o, 32); }
    if (lane == 0) { red[w] = sum; red[w + 8] = sq; }
    __syncthreads();
    if (w == 0) {
        float s2 = (lane < 8) ? red[lane] : 0.f;
        float q2 = (lane < 8) ? red[lane + 8] : 0.f;
        for (int o = 4; o > 0; o >>= 1) { s2 += __shfl_xor(s2, o, 32); q2 += __shfl_xor(q2, o, 32); }
        if (lane == 0) { red[0] = s2; red[1] = q2; }
    }
    __syncthreads();
    float mean = red[0] * (1.0f / 1024.0f);
    float var  = red[1] * (1.0f / 1024.0f) - mean * mean;
    float rstd = rsqrtf(var + 1e-5f);

    float4 gg = *(const float4*)&g[c0];
    float4 bb = *(const float4*)&b[c0];
    float y0 = (v.x - mean) * rstd * gg.x + bb.x;
    float y1 = (v.y - mean) * rstd * gg.y + bb.y;
    float y2 = (v.z - mean) * rstd * gg.z + bb.z;
    float y3 = (v.w - mean) * rstd * gg.w + bb.w;
    uint2 pk = make_uint2(pack2bf(y0, y1), pack2bf(y2, y3));
    *(uint2*)&kvin[(size_t)r * 1024 + c0] = pk;
    if (isl) *(uint2*)&lnout[((size_t)bt * 64 + (s - 576)) * 1024 + c0] = pk;
}

// ---------- bf16 GEMM: C[M,N] = A[M,K] @ B[K,N], fp32 accumulate ----------
// block tile 128x64, 8 waves (4x2), each wave 32x32 (2x2 WMMA tiles), K step 32,
// double-buffered: async-DMA A tile, register-staged transposed B tile.
template <bool OUT_BF16>
__global__ __launch_bounds__(256) void gemm_bf16_kernel(
    const ushort_t* __restrict__ A, const ushort_t* __restrict__ B,
    void* __restrict__ Cv, int M, int N, int K) {
    __shared__ __align__(16) ushort_t As[2][128 * 32];
    __shared__ __align__(16) ushort_t Bs[2][64 * 32];   // transposed: [n][k]

    int t = threadIdx.x, lane = t & 31, w = t >> 5;
    int mblk = blockIdx.y * 128, nblk = blockIdx.x * 64;
    int mw = (w >> 1) * 32, nw = (w & 1) * 32;
    int kr = t >> 3, nc = (t & 7) * 8;                  // B staging coords

    auto stageA = [&](int buf, int kb) {
#pragma unroll
        for (int i = 0; i < 2; ++i) {
            int e = t + i * 256;                         // 0..511
            int r = e >> 2, c = (e & 3) * 8;
            async_ld_b128(&As[buf][r * 32 + c], &A[(size_t)(mblk + r) * K + kb + c]);
        }
    };
    auto loadB = [&](int kb) -> uint4 {
        return *(const uint4*)&B[(size_t)(kb + kr) * N + nblk + nc];
    };
    auto scatterB = [&](int buf, uint4 d) {
        const ushort_t* ds = (const ushort_t*)&d;
#pragma unroll
        for (int j = 0; j < 8; ++j) Bs[buf][(nc + j) * 32 + kr] = ds[j];
    };

    v8f acc[2][2] = {};
    int nk = K >> 5;

    // prologue: stage tile 0
    stageA(0, 0);
    scatterB(0, loadB(0));
    async_wait0();
    __syncthreads();

    for (int ki = 0; ki < nk; ++ki) {
        int cur = ki & 1, nxt = cur ^ 1;
        bool more = (ki + 1 < nk);
        uint4 bregs = {};
        if (more) {
            int kb1 = (ki + 1) << 5;
            stageA(nxt, kb1);                            // async DMA, no VGPR traffic
            bregs = loadB(kb1);
            if (ki + 2 < nk)                             // speculative prefetch (global_prefetch_b8)
                __builtin_prefetch(&B[(size_t)(((ki + 2) << 5) + kr) * N + nblk + nc], 0, 0);
        }
#pragma unroll
        for (int tm = 0; tm < 2; ++tm) {
            v16bf a = load_frag_a(As[cur], 32, mw + tm * 16, lane, 0);
#pragma unroll
            for (int tn = 0; tn < 2; ++tn) {
                v16bf b = load_frag_b(Bs[cur], 32, nw + tn * 16, lane, 0);
                acc[tm][tn] = __builtin_amdgcn_wmma_f32_16x16x32_bf16(
                    false, a, false, b, (short)0, acc[tm][tn], false, false);
            }
        }
        if (more) scatterB(nxt, bregs);
        async_wait0();
        __syncthreads();
    }

#pragma unroll
    for (int tm = 0; tm < 2; ++tm)
#pragma unroll
        for (int tn = 0; tn < 2; ++tn) {
            int col = nblk + nw + tn * 16 + (lane & 15);
            int row0 = mblk + mw + tm * 16 + ((lane >> 4) & 1) * 8;
#pragma unroll
            for (int vv = 0; vv < 8; vv += 2) {
                float v0 = acc[tm][tn][vv], v1 = acc[tm][tn][vv + 1];
                if (OUT_BF16) {
                    uint32 pk = pack2bf(v0, v1);
                    ((ushort_t*)Cv)[(size_t)(row0 + vv) * N + col]     = (ushort_t)pk;
                    ((ushort_t*)Cv)[(size_t)(row0 + vv + 1) * N + col] = (ushort_t)(pk >> 16);
                } else {
                    ((float*)Cv)[(size_t)(row0 + vv) * N + col]     = v0;
                    ((float*)Cv)[(size_t)(row0 + vv + 1) * N + col] = v1;
                }
            }
        }
}

// ---------- attention: per (head, bt) block ----------
// q: [4096][1024] bf16 ; kv: [40960][2048] bf16 (k cols 0..1023, v cols 1024..2047)
__global__ __launch_bounds__(256) void attn_kernel(
    const ushort_t* __restrict__ q, const ushort_t* __restrict__ kv,
    const int* __restrict__ mask, ushort_t* __restrict__ ao) {
    const int h  = blockIdx.x;   // 16
    const int bt = blockIdx.y;   // 64
    __shared__ __align__(16) ushort_t q_s[64 * 64];    //  8 KB
    __shared__ __align__(16) ushort_t c_s[160 * 64];   // 20 KB: k chunk [s][d] or vT chunk [d][s]
    __shared__ __align__(16) ushort_t p_s[64 * 640];   // 80 KB: sim / probs (bf16)
    __shared__ float rinv[64];

    int t = threadIdx.x, lane = t & 31, w = t >> 5;
    const float scale = 0.125f;  // 64^-0.5
    const size_t qbase  = (size_t)bt * 64 * 1024 + h * 64;
    const size_t kvbase = (size_t)bt * 640 * 2048 + h * 64;

    // load q tile (64x64) async
    for (int e = t; e < 512; e += 256) {
        int r = e >> 3, c = (e & 7) * 8;
        async_ld_b128(&q_s[r * 64 + c], &q[qbase + (size_t)r * 1024 + c]);
    }

    // ---- sim = q @ k^T (in 4 key chunks of 160) ----
    for (int cc = 0; cc < 4; ++cc) {
        __syncthreads();
        for (int e = t; e < 1280; e += 256) {                 // k chunk 160x64, async
            int r = e >> 3, c = (e & 7) * 8;
            async_ld_b128(&c_s[r * 64 + c], &kv[kvbase + (size_t)(cc * 160 + r) * 2048 + c]);
        }
        async_wait0();
        __syncthreads();
#pragma unroll
        for (int i = 0; i < 5; ++i) {                         // 40 (mt,nt) pairs / 8 waves
            int p = w + i * 8;
            int mt = p / 10, nt = p % 10;
            v8f acc = {};
#pragma unroll
            for (int kc = 0; kc < 2; ++kc) {                  // dh = 64 = 2 * 32
                v16bf a = load_frag_a(q_s, 64, mt * 16, lane, kc * 32);
                v16bf b = load_frag_b(c_s, 64, nt * 16, lane, kc * 32);
                acc = __builtin_amdgcn_wmma_f32_16x16x32_bf16(false, a, false, b, (short)0, acc, false, false);
            }
            int col = cc * 160 + nt * 16 + (lane & 15);
            int row0 = mt * 16 + ((lane >> 4) & 1) * 8;
#pragma unroll
            for (int vv = 0; vv < 8; vv += 2) {
                uint32 pk = pack2bf(acc[vv], acc[vv + 1]);
                p_s[(row0 + vv) * 640 + col]     = (ushort_t)pk;
                p_s[(row0 + vv + 1) * 640 + col] = (ushort_t)(pk >> 16);
            }
        }
    }
    __syncthreads();

    // ---- softmax (wave per 8 rows); column-pair processing; scale+mask folded ----
    for (int rr = 0; rr < 8; ++rr) {
        int row = w * 8 + rr;
        float vals[20];
        float mx = -3.0e38f;
#pragma unroll
        for (int i = 0; i < 10; ++i) {
            int j = 2 * lane + i * 64;                       // 576 = 9*64: i<9 -> masked region
            uint32 pk = *(const uint32*)&p_s[row * 640 + j];
            float s0 = bf2f((ushort_t)(pk & 0xFFFFu)) * scale;
            float s1 = bf2f((ushort_t)(pk >> 16)) * scale;
            if (i < 9) {
                if (mask[bt * 576 + j] == 0)     s0 = -1.0e30f;
                if (mask[bt * 576 + j + 1] == 0) s1 = -1.0e30f;
            }
            vals[2 * i] = s0; vals[2 * i + 1] = s1;
            mx = fmaxf(mx, fmaxf(s0, s1));
        }
        for (int o = 16; o > 0; o >>= 1) mx = fmaxf(mx, __shfl_xor(mx, o, 32));
        float sum = 0.f;
#pragma unroll
        for (int i = 0; i < 10; ++i) {
            int j = 2 * lane + i * 64;
            float e0 = __expf(vals[2 * i] - mx);
            float e1 = __expf(vals[2 * i + 1] - mx);
            sum += e0 + e1;
            *(uint32*)&p_s[row * 640 + j] = pack2bf(e0, e1);
        }
        for (int o = 16; o > 0; o >>= 1) sum += __shfl_xor(sum, o, 32);
        if (lane == 0) rinv[row] = 1.0f / sum;
    }

    // ---- out = p @ v (4 key chunks; V staged transposed vT[d][s], pitch 160) ----
    v8f oacc[2] = {};
    for (int cc = 0; cc < 4; ++cc) {
        __syncthreads();
        for (int e = t; e < 10240; e += 256) {                // transpose scatter (VGPR path)
            int s = e >> 6, d = e & 63;
            c_s[d * 160 + s] = kv[kvbase + (size_t)(cc * 160 + s) * 2048 + 1024 + d];
        }
        __syncthreads();
#pragma unroll
        for (int i = 0; i < 2; ++i) {                         // 16 (mt,nt) pairs / 8 waves
            int p = w * 2 + i;
            int mt = p >> 2, nt = p & 3;
#pragma unroll
            for (int ks = 0; ks < 5; ++ks) {                  // 160 = 5 * 32
                v16bf a = load_frag_a(p_s, 640, mt * 16, lane, cc * 160 + ks * 32);
                v16bf b = load_frag_b(c_s, 160, nt * 16, lane, ks * 32);
                oacc[i] = __builtin_amdgcn_wmma_f32_16x16x32_bf16(false, a, false, b, (short)0, oacc[i], false, false);
            }
        }
    }

#pragma unroll
    for (int i = 0; i < 2; ++i) {
        int p = w * 2 + i;
        int mt = p >> 2, nt = p & 3;
        int col = nt * 16 + (lane & 15);
        int row0 = mt * 16 + ((lane >> 4) & 1) * 8;
#pragma unroll
        for (int vv = 0; vv < 8; vv += 2) {
            uint32 pk = pack2bf(oacc[i][vv] * rinv[row0 + vv],
                                oacc[i][vv + 1] * rinv[row0 + vv + 1]);
            ao[qbase + (size_t)(row0 + vv) * 1024 + col]     = (ushort_t)pk;
            ao[qbase + (size_t)(row0 + vv + 1) * 1024 + col] = (ushort_t)(pk >> 16);
        }
    }
}

// ---------- launcher ----------
extern "C" void kernel_launch(void* const* d_in, const int* in_sizes, int n_in,
                              void* d_out, int out_size, void* d_ws, size_t ws_size,
                              hipStream_t stream) {
    const float* x   = (const float*)d_in[0];
    const float* lat = (const float*)d_in[1];
    const int*   msk = (const int*)  d_in[2];
    const float* gm  = (const float*)d_in[3];
    const float* bm  = (const float*)d_in[4];
    const float* gl  = (const float*)d_in[5];
    const float* bl  = (const float*)d_in[6];
    const float* Wq  = (const float*)d_in[7];
    const float* Wkv = (const float*)d_in[8];
    const float* Wo  = (const float*)d_in[9];
    float* out = (float*)d_out;

    size_t off = 0;
    auto alloc = [&](size_t bytes) -> void* {
        void* p = (char*)d_ws + off;
        off += (bytes + 255) & ~(size_t)255;
        return p;
    };
    ushort_t* kvin = (ushort_t*)alloc((size_t)40960 * 1024 * 2);
    ushort_t* lnb  = (ushort_t*)alloc((size_t)4096  * 1024 * 2);
    ushort_t* qb   = (ushort_t*)alloc((size_t)4096  * 1024 * 2);
    ushort_t* kvb  = (ushort_t*)alloc((size_t)40960 * 2048 * 2);
    ushort_t* aob  = (ushort_t*)alloc((size_t)4096  * 1024 * 2);
    ushort_t* wqb  = (ushort_t*)alloc((size_t)1024  * 1024 * 2);
    ushort_t* wkvb = (ushort_t*)alloc((size_t)1024  * 2048 * 2);
    ushort_t* wob  = (ushort_t*)alloc((size_t)1024  * 1024 * 2);
    (void)ws_size; (void)in_sizes; (void)n_in; (void)out_size;

    // 1) weights -> bf16 (2 elems/thread)
    cvt_bf16_kernel<<<(1024 * 1024 / 2 + 255) / 256, 256, 0, stream>>>(Wq,  wqb,  1024 * 1024);
    cvt_bf16_kernel<<<(1024 * 2048 / 2 + 255) / 256, 256, 0, stream>>>(Wkv, wkvb, 1024 * 2048);
    cvt_bf16_kernel<<<(1024 * 1024 / 2 + 255) / 256, 256, 0, stream>>>(Wo,  wob,  1024 * 1024);

    // 2) layernorms -> bf16 kv_input (+ contiguous ln)
    ln_kernel<<<40960, 256, 0, stream>>>(x, lat, gm, bm, gl, bl, kvin, lnb);

    // 3) q = ln @ Wq   (4096 x 1024 x 1024)
    gemm_bf16_kernel<true><<<dim3(1024 / 64, 4096 / 128), 256, 0, stream>>>(lnb, wqb, qb, 4096, 1024, 1024);

    // 4) kv = kv_input @ Wkv   (40960 x 2048 x 1024)
    gemm_bf16_kernel<true><<<dim3(2048 / 64, 40960 / 128), 256, 0, stream>>>(kvin, wkvb, kvb, 40960, 2048, 1024);

    // 5) attention per (head, bt)
    attn_kernel<<<dim3(16, 64), 256, 0, stream>>>(qb, kvb, msk, aob);

    // 6) out = ao @ Wo   (4096 x 1024 x 1024) -> fp32
    gemm_bf16_kernel<false><<<dim3(1024 / 64, 4096 / 128), 256, 0, stream>>>(aob, wob, out, 4096, 1024, 1024);
}
